// LSTMModel_54107997995130
// MI455X (gfx1250) — matrix operand order repaired and checked by reading, not compiled, not run on hardware
//
#include <hip/hip_runtime.h>

typedef __attribute__((ext_vector_type(16))) _Float16     v16h;
typedef __attribute__((ext_vector_type(8)))  float        v8f;
typedef __attribute__((ext_vector_type(4)))  unsigned int u32x4;

#define HID            50
#define HP             64                 // per-gate padded width
#define NP             256                // 4 gates * HP
#define KP             128                // [input(64) | h(64)] in the weight image
#define T_SEQ          512
#define BATCH          2048
#define ROWS           32                 // rows per block = one wave, two M-tiles
#define NBLOCKS        (BATCH / ROWS)     // 64 blocks

#define W_ELEMS    (3 * NP * KP)          // f16 weight image (192 KB)
#define BIAS_ELEMS (3 * NP)               // f32 fused bias
#define HBUF_ELEMS (3 * ROWS * HP)        // f16: h1,h2,h3 slots
#define XS_BYTES   (ROWS * 4)             // f32 x staging
#define SMEM_BYTES (W_ELEMS * 2 + BIAS_ELEMS * 4 + HBUF_ELEMS * 2 + XS_BYTES)

#if __has_builtin(__builtin_amdgcn_tanhf)
__device__ __forceinline__ float tanh_f(float v) { return __builtin_amdgcn_tanhf(v); }
__device__ __forceinline__ float sigm(float v)   { return __builtin_amdgcn_tanhf(v * 0.5f) * 0.5f + 0.5f; }
#else
__device__ __forceinline__ float tanh_f(float v) {
    return 2.0f * __builtin_amdgcn_rcpf(1.0f + __expf(-2.0f * v)) - 1.0f;
}
__device__ __forceinline__ float sigm(float v) {
    return __builtin_amdgcn_rcpf(1.0f + __expf(-v));
}
#endif

__device__ __forceinline__ v8f splat8(float v) {
    v8f r = {v, v, v, v, v, v, v, v};
    return r;
}

// A fragment (16x32 f16, row-major LDS, row stride HP): lane&15=M, lane>>4=K subgroup
__device__ __forceinline__ v16h load_a(const _Float16* buf, int kbase, int row, int hi) {
    union { v16h v; u32x4 q[2]; } u;
    const _Float16* p = buf + row * HP + kbase + hi * 8;
    u.q[0] = *(const u32x4*)p;
    u.q[1] = *(const u32x4*)(p + 16);
    return u.v;
}

// B fragment (32x16 f16): weights n-major [NP][KP]; lane&15=N, lane>>4=K subgroup
__device__ __forceinline__ v16h load_b(const _Float16* w, int n, int kbase, int hi) {
    union { v16h v; u32x4 q[2]; } u;
    const _Float16* p = w + n * KP + kbase + hi * 8;
    u.q[0] = *(const u32x4*)p;
    u.q[1] = *(const u32x4*)(p + 16);
    return u.v;
}

__device__ __forceinline__ v8f wmma_f16(v16h a, v16h b, v8f c) {
    return __builtin_amdgcn_wmma_f32_16x16x32_f16(false, a, false, b, (short)0, c, false, false);
}

__global__ void __launch_bounds__(32)
lstm3_fused_persistent(const float* __restrict__ x,
                       const float* __restrict__ Wih1, const float* __restrict__ Whh1,
                       const float* __restrict__ bih1, const float* __restrict__ bhh1,
                       const float* __restrict__ Wih2, const float* __restrict__ Whh2,
                       const float* __restrict__ bih2, const float* __restrict__ bhh2,
                       const float* __restrict__ Wih3, const float* __restrict__ Whh3,
                       const float* __restrict__ bih3, const float* __restrict__ bhh3,
                       const float* __restrict__ fcW,  const float* __restrict__ fcb,
                       float* __restrict__ out)
{
    extern __shared__ char smem[];
    _Float16* Wc = (_Float16*)smem;                                   // [3][NP][KP]
    float*    Bc = (float*)(smem + W_ELEMS * 2);                      // [3][NP]
    _Float16* Hb = (_Float16*)(smem + W_ELEMS * 2 + BIAS_ELEMS * 4);  // [3][ROWS][HP]
    float*    Xs = (float*)(smem + W_ELEMS * 2 + BIAS_ELEMS * 4 + HBUF_ELEMS * 2);

    const int lane = threadIdx.x & 31;
    const int l15  = lane & 15;
    const int hi   = lane >> 4;

    // ---- LDS init: padded, pre-fused weight image + bias (single wave) ----
    for (int idx = lane; idx < W_ELEMS; idx += 32) {
        const int l  = idx / (NP * KP);
        const int r0 = idx - l * NP * KP;
        const int n  = r0 / KP;
        const int k  = r0 - n * KP;
        const int gi = n >> 6, nr = n & 63;
        float w = 0.0f;
        if (nr < HID) {
            const int no = gi * HID + nr;
            if (k < HP) {                       // input-projection half
                if (l == 0) { if (k == 0) w = Wih1[no]; }
                else if (k < HID) {
                    const float* Wih = (l == 1) ? Wih2 : Wih3;
                    w = Wih[no * HID + k];
                }
            } else {                            // recurrent half
                const int kr = k - HP;
                if (kr < HID) {
                    const float* Whh = (l == 0) ? Whh1 : ((l == 1) ? Whh2 : Whh3);
                    w = Whh[no * HID + kr];
                }
            }
        }
        Wc[idx] = (_Float16)w;
    }
    for (int idx = lane; idx < BIAS_ELEMS; idx += 32) {
        const int l = idx / NP, n = idx - l * NP;
        const int gi = n >> 6, nr = n & 63;
        float b = 0.0f;
        if (nr < HID) {
            const float* bih = (l == 0) ? bih1 : ((l == 1) ? bih2 : bih3);
            const float* bhh = (l == 0) ? bhh1 : ((l == 1) ? bhh2 : bhh3);
            b = bih[gi * HID + nr] + bhh[gi * HID + nr];
        }
        Bc[idx] = b;
    }
    for (int idx = lane; idx < HBUF_ELEMS; idx += 32) Hb[idx] = (_Float16)0.0f;
    __syncthreads();

    // ---- loop-invariant per-lane constants -> registers (1 wave/SIMD: big VGPR budget)
    float bias_r[3][16];   // [layer][tile tt=jt*4+gi] at column gi*HP+jt*16+l15
    float w1f[16];         // layer-0 input weight column (rank-1 input projection)
#pragma unroll
    for (int l = 0; l < 3; ++l)
#pragma unroll
        for (int tt = 0; tt < 16; ++tt)
            bias_r[l][tt] = Bc[l * NP + (tt & 3) * HP + (tt >> 2) * 16 + l15];
#pragma unroll
    for (int tt = 0; tt < 16; ++tt)
        w1f[tt] = (float)Wc[((tt & 3) * HP + (tt >> 2) * 16 + l15) * KP];

    const int rowbase = blockIdx.x * ROWS;

    v8f c_st[3][2][4];
#pragma unroll
    for (int l = 0; l < 3; ++l)
#pragma unroll
        for (int m = 0; m < 2; ++m)
#pragma unroll
            for (int jt = 0; jt < 4; ++jt) c_st[l][m][jt] = splat8(0.0f);

    // layer-0 recurrent A fragments (rotating; slot0 is zeroed -> h1(-1)=0)
    v16h ar0[2][2];
#pragma unroll
    for (int m = 0; m < 2; ++m) {
        ar0[m][0] = load_a(Hb, 0,  m * 16 + l15, hi);
        ar0[m][1] = load_a(Hb, 32, m * 16 + l15, hi);
    }

    float xv = x[(rowbase + lane) * T_SEQ];   // t=0 prefetched

    for (int t = 0; t < T_SEQ; ++t) {
        Xs[lane] = xv;
        if (t + 1 < T_SEQ) xv = x[(rowbase + lane) * T_SEQ + t + 1];

        // broadcast x per row: xrow[m][r] = x[rowbase + m*16 + 8*hi + r]
        v8f xrow[2];
#pragma unroll
        for (int m = 0; m < 2; ++m) {
            union { v8f v; u32x4 q[2]; } u;
            const float* p = Xs + m * 16 + hi * 8;
            u.q[0] = *(const u32x4*)p;
            u.q[1] = *(const u32x4*)(p + 4);
            xrow[m] = u.v;
        }

        // ================= layer 0: rank-1 input + K=64 recurrent GEMM ============
        {
            _Float16* own = Hb;                       // slot0 = h1
            const _Float16* wl = Wc;                  // layer 0 image (recurrent half)
            v16h bb0[2][2];
            bb0[0][0] = load_b(wl, l15, 64, hi);
            bb0[0][1] = load_b(wl, l15, 96, hi);

#pragma unroll
            for (int jt = 0; jt < 4; ++jt) {
                v8f acc[2][4];
#pragma unroll
                for (int gi = 0; gi < 4; ++gi) {
                    const int tt  = jt * 4 + gi;
                    const int cur = tt & 1;
                    if (tt + 1 < 16) {
                        const int nn = ((tt + 1) & 3) * HP + ((tt + 1) >> 2) * 16 + l15;
                        bb0[cur ^ 1][0] = load_b(wl, nn, 64, hi);
                        bb0[cur ^ 1][1] = load_b(wl, nn, 96, hi);
                    }
                    acc[0][gi] = xrow[0] * splat8(w1f[tt]) + splat8(bias_r[0][tt]);
                    acc[1][gi] = xrow[1] * splat8(w1f[tt]) + splat8(bias_r[0][tt]);
#pragma unroll
                    for (int kc = 0; kc < 2; ++kc) {
                        acc[0][gi] = wmma_f16(ar0[0][kc], bb0[cur][kc], acc[0][gi]);
                        acc[1][gi] = wmma_f16(ar0[1][kc], bb0[cur][kc], acc[1][gi]);
                    }
                }
#pragma unroll
                for (int m = 0; m < 2; ++m) {
                    v8f c_old = c_st[0][m][jt];
#pragma unroll
                    for (int r = 0; r < 8; ++r) {
                        const float iv = sigm(acc[m][0][r]);
                        const float fv = sigm(acc[m][1][r]);
                        const float gv = tanh_f(acc[m][2][r]);
                        const float ov = sigm(acc[m][3][r]);
                        const float c  = fv * c_old[r] + iv * gv;
                        c_old[r] = c;
                        const float h = ov * tanh_f(c);
                        own[(m * 16 + r + 8 * hi) * HP + jt * 16 + l15] = (_Float16)h;
                    }
                    c_st[0][m][jt] = c_old;
                }
            }
            // rotate: prefetch next-t recurrent A now that h1(t) is final;
            // latency hidden under layers 1-2
#pragma unroll
            for (int m = 0; m < 2; ++m) {
                ar0[m][0] = load_a(Hb, 0,  m * 16 + l15, hi);
                ar0[m][1] = load_a(Hb, 32, m * 16 + l15, hi);
            }
        }

        // ================= layers 1,2: full K=128 GEMM ============================
#pragma unroll
        for (int li = 0; li < 2; ++li) {
            const int l = li + 1;
            const _Float16* inSlot = Hb + li * (ROWS * HP);
            _Float16*       own    = Hb + (li + 1) * (ROWS * HP);
            const _Float16* wl     = Wc + l * (NP * KP);

            v16h a[2][4];
#pragma unroll
            for (int m = 0; m < 2; ++m) {
                const int row = m * 16 + l15;
                a[m][0] = load_a(inSlot, 0,  row, hi);
                a[m][1] = load_a(inSlot, 32, row, hi);
                a[m][2] = load_a(own,    0,  row, hi);
                a[m][3] = load_a(own,    32, row, hi);
            }

            v16h bb[2][4];
#pragma unroll
            for (int kc = 0; kc < 4; ++kc) bb[0][kc] = load_b(wl, l15, kc * 32, hi);

#pragma unroll
            for (int jt = 0; jt < 4; ++jt) {
                v8f acc[2][4];
#pragma unroll
                for (int gi = 0; gi < 4; ++gi) {
                    const int tt  = jt * 4 + gi;
                    const int cur = tt & 1;
                    if (tt + 1 < 16) {
                        const int nn = ((tt + 1) & 3) * HP + ((tt + 1) >> 2) * 16 + l15;
#pragma unroll
                        for (int kc = 0; kc < 4; ++kc)
                            bb[cur ^ 1][kc] = load_b(wl, nn, kc * 32, hi);
                    }
                    acc[0][gi] = splat8(bias_r[l][tt]);
                    acc[1][gi] = splat8(bias_r[l][tt]);
#pragma unroll
                    for (int kc = 0; kc < 4; ++kc) {
                        acc[0][gi] = wmma_f16(a[0][kc], bb[cur][kc], acc[0][gi]);
                        acc[1][gi] = wmma_f16(a[1][kc], bb[cur][kc], acc[1][gi]);
                    }
                }
#pragma unroll
                for (int m = 0; m < 2; ++m) {
                    v8f c_old = c_st[l][m][jt];
#pragma unroll
                    for (int r = 0; r < 8; ++r) {
                        const float iv = sigm(acc[m][0][r]);
                        const float fv = sigm(acc[m][1][r]);
                        const float gv = tanh_f(acc[m][2][r]);
                        const float ov = sigm(acc[m][3][r]);
                        const float c  = fv * c_old[r] + iv * gv;
                        c_old[r] = c;
                        const float h = ov * tanh_f(c);
                        own[(m * 16 + r + 8 * hi) * HP + jt * 16 + l15] = (_Float16)h;
                    }
                    c_st[l][m][jt] = c_old;
                }
            }
        }
    }

    // final FC on h3(T-1): all 32 lanes, one batch row each
    {
        const _Float16* h3 = Hb + 2 * (ROWS * HP) + lane * HP;
        float s = fcb[0];
#pragma unroll
        for (int j = 0; j < HID; ++j) s += (float)h3[j] * fcW[j];
        out[rowbase + lane] = s;
    }
}

extern "C" void kernel_launch(void* const* d_in, const int* in_sizes, int n_in,
                              void* d_out, int out_size, void* d_ws, size_t ws_size,
                              hipStream_t stream) {
    (void)in_sizes; (void)n_in; (void)d_ws; (void)ws_size; (void)out_size;
    const float* x    = (const float*)d_in[0];
    const float* Wih1 = (const float*)d_in[1];
    const float* Whh1 = (const float*)d_in[2];
    const float* bih1 = (const float*)d_in[3];
    const float* bhh1 = (const float*)d_in[4];
    const float* Wih2 = (const float*)d_in[5];
    const float* Whh2 = (const float*)d_in[6];
    const float* bih2 = (const float*)d_in[7];
    const float* bhh2 = (const float*)d_in[8];
    const float* Wih3 = (const float*)d_in[9];
    const float* Whh3 = (const float*)d_in[10];
    const float* bih3 = (const float*)d_in[11];
    const float* bhh3 = (const float*)d_in[12];
    const float* fcW  = (const float*)d_in[13];
    const float* fcb  = (const float*)d_in[14];
    float* out = (float*)d_out;

    hipFuncSetAttribute((const void*)lstm3_fused_persistent,
                        hipFuncAttributeMaxDynamicSharedMemorySize, SMEM_BYTES);

    lstm3_fused_persistent<<<NBLOCKS, 32, SMEM_BYTES, stream>>>(
        x, Wih1, Whh1, bih1, bhh1, Wih2, Whh2, bih2, bhh2,
        Wih3, Whh3, bih3, bhh3, fcW, fcb, out);
}
